// GNNModel_89704686944905
// MI455X (gfx1250) — matrix operand order repaired and checked by reading, not compiled, not run on hardware
//
#include <hip/hip_runtime.h>
#include <math.h>

#define N_NODES 100000
#define N_EDGES 1000000
#define N_GRAPHS 100
#define IN_F 128
#define HID 64
#define N_CLS 40
#define BN_EPS 1e-5f

typedef __attribute__((ext_vector_type(16))) _Float16 v16h;
typedef __attribute__((ext_vector_type(8)))  float    v8f;
typedef __attribute__((ext_vector_type(4)))  float    v4f;

__device__ __forceinline__ float gelu_exact(float x) {
  return 0.5f * x * (1.0f + erff(x * 0.70710678118654752440f));
}

// ---------------- utility ----------------
__global__ void zero_f32(float* __restrict__ p, long long n) {
  long long i = (long long)blockIdx.x * blockDim.x + threadIdx.x;
  if (i < n) p[i] = 0.0f;
}

__global__ void deg_kernel(const int* __restrict__ src, const int* __restrict__ dst,
                           float* degOut, float* degIn, int nE) {
  int e = blockIdx.x * blockDim.x + threadIdx.x;
  if (e < nE) {
    atomicAdd(&degOut[src[e]], 1.0f);
    atomicAdd(&degIn [dst[e]], 1.0f);
  }
}

__global__ void dn_finalize(float* dOut, float* dIn, int n) {
  int i = blockIdx.x * blockDim.x + threadIdx.x;
  if (i < n) {
    float a = dOut[i]; dOut[i] = (a > 0.0f) ? 1.0f / sqrtf(a) : 1.0f;
    float b = dIn[i];  dIn[i]  = (b > 0.0f) ? 1.0f / sqrtf(b) : 1.0f;
  }
}

// Pack W [K, 64] fp32 row-major into CDNA5 WMMA f16 B-fragments.
// Fragment layout (per 16x16x32 B tile): lane L holds column N = L%16;
// lanes 0-15 hold K 0..7,16..23 of the k-step, lanes 16-31 hold 8..15,24..31.
__global__ void pack_w(const float* __restrict__ W, _Float16* __restrict__ out, int K) {
  int t = blockIdx.x * blockDim.x + threadIdx.x;
  int total = (K / 32) * 4 * 32;   // (ksteps * ntiles) fragments, 32 lanes each
  if (t >= total) return;
  int lane = t & 31;
  int frag = t >> 5;
  int nt   = frag & 3;
  int ks   = frag >> 2;
  int n  = nt * 16 + (lane & 15);
  int kb = ks * 32 + ((lane >> 4) ? 8 : 0);
#pragma unroll
  for (int hi = 0; hi < 16; ++hi)
    out[(size_t)t * 16 + hi] = (_Float16)W[(size_t)(kb + (hi < 8 ? hi : hi + 8)) * HID + n];
}

// Y[N,64] = (diag(rowScale) * X[N,K]) @ W[K,64], via v_wmma_f32_16x16x32_f16.
// One wave computes a 16-row x 64-col tile (4 accumulators).
__global__ void gemm_wmma(const float* __restrict__ X, const _Float16* __restrict__ Bp,
                          const float* __restrict__ rowScale,
                          float* __restrict__ Y, int N, int K) {
  int lane = threadIdx.x & 31;
  int wave = threadIdx.x >> 5;
  int row0 = (blockIdx.x * 4 + wave) * 16;
  if (row0 >= N) return;                       // wave-uniform: EXEC stays all-ones
  int m = lane & 15;
  int r = row0 + m;
  float sc = rowScale ? rowScale[r] : 1.0f;
  const float* xrow = X + (size_t)r * K;
  int hiSel = (lane >> 4) << 3;                // 0 for lanes 0-15, 8 for lanes 16-31

  v8f acc[4] = {};
  int ksteps = K >> 5;
  for (int ks = 0; ks < ksteps; ++ks) {
    // A fragment: halves 0..7 = K kLo..kLo+7, halves 8..15 = K kLo+16..kLo+23
    int kLo = (ks << 5) + hiSel;
    v4f x0 = ((const v4f*)(xrow + kLo))[0];
    v4f x1 = ((const v4f*)(xrow + kLo))[1];
    v4f x2 = ((const v4f*)(xrow + kLo + 16))[0];
    v4f x3 = ((const v4f*)(xrow + kLo + 16))[1];
    v16h a;
#pragma unroll
    for (int i = 0; i < 4; ++i) {
      a[i]      = (_Float16)(x0[i] * sc);
      a[4 + i]  = (_Float16)(x1[i] * sc);
      a[8 + i]  = (_Float16)(x2[i] * sc);
      a[12 + i] = (_Float16)(x3[i] * sc);
    }
    const _Float16* bbase = Bp + ((size_t)(ks * 4) * 32 + lane) * 16;
#pragma unroll
    for (int nt = 0; nt < 4; ++nt) {
      v16h b = *(const v16h*)(bbase + (size_t)nt * 32 * 16);
      acc[nt] = __builtin_amdgcn_wmma_f32_16x16x32_f16(
          false, a, false, b, (short)0, acc[nt], false, false);
    }
  }
  // C/D layout: VGPR v -> M = v (lanes 0-15) / M = v+8 (lanes 16-31)
  int outr = row0 + hiSel;
  int outc = lane & 15;
#pragma unroll
  for (int nt = 0; nt < 4; ++nt)
#pragma unroll
    for (int v = 0; v < 8; ++v)
      Y[(size_t)(outr + v) * HID + nt * 16 + outc] = acc[nt][v];
}

// agg[dst[e], c] += h[src[e], c] * (rowScale ? rowScale[src[e]] : 1)
__global__ void scatter_add(const float* __restrict__ h, const int* __restrict__ src,
                            const int* __restrict__ dst, const float* __restrict__ rowScale,
                            float* __restrict__ agg, int nE) {
  long long idx = (long long)blockIdx.x * blockDim.x + threadIdx.x;
  if (idx >= (long long)nE * HID) return;
  int e = (int)(idx >> 6);
  int c = (int)(idx & 63);
  int s = src[e];
  float v = h[(size_t)s * HID + c];
  if (rowScale) v *= rowScale[s];
  atomicAdd(&agg[(size_t)dst[e] * HID + c], v);
}

// out = gelu(agg*dnIn + bg) + gelu(res + br)
__global__ void epilogue_k(const float* __restrict__ agg, const float* __restrict__ res,
                           const float* __restrict__ dnIn,
                           const float* __restrict__ bg, const float* __restrict__ br,
                           float* __restrict__ out, int N) {
  long long idx = (long long)blockIdx.x * blockDim.x + threadIdx.x;
  if (idx >= (long long)N * HID) return;
  int r = (int)(idx >> 6);
  int c = (int)(idx & 63);
  float a = agg[idx];
  if (dnIn) a *= dnIn[r];
  out[idx] = gelu_exact(a + bg[c]) + gelu_exact(res[idx] + br[c]);
}

// per-channel sum / sumsq, 64 threads per block (one per channel)
__global__ void bn_stats(const float* __restrict__ x, float* __restrict__ sums, int N) {
  int c = threadIdx.x;
  float s = 0.0f, q = 0.0f;
  for (int r = blockIdx.x; r < N; r += gridDim.x) {
    float v = x[(size_t)r * HID + c];
    s += v; q += v * v;
  }
  atomicAdd(&sums[c], s);
  atomicAdd(&sums[64 + c], q);
}

__global__ void bn_finalize(const float* __restrict__ sums, const float* __restrict__ g,
                            const float* __restrict__ be, float* __restrict__ ss, int N) {
  int c = threadIdx.x;
  if (c >= HID) return;
  float invN = 1.0f / (float)N;
  float mean = sums[c] * invN;
  float var  = sums[64 + c] * invN - mean * mean;   // biased variance
  float scale = g[c] * rsqrtf(var + BN_EPS);
  ss[c]      = scale;
  ss[64 + c] = be[c] - mean * scale;
}

__global__ void bn_apply(const float* __restrict__ x, const float* __restrict__ ss,
                         float* __restrict__ y, int N) {
  long long idx = (long long)blockIdx.x * blockDim.x + threadIdx.x;
  if (idx >= (long long)N * HID) return;
  int c = (int)(idx & 63);
  y[idx] = x[idx] * ss[c] + ss[64 + c];
}

// out[g, j] = h[offsets[g], :] @ Wl[:, j] + bl[j]
__global__ void classify(const float* __restrict__ h, const int* __restrict__ offsets,
                         const float* __restrict__ Wl, const float* __restrict__ bl,
                         float* __restrict__ out) {
  int g = blockIdx.x;
  int j = threadIdx.x;
  if (j >= N_CLS) return;
  const float* row = h + (size_t)offsets[g] * HID;
  float acc = bl[j];
#pragma unroll 8
  for (int c = 0; c < HID; ++c) acc += row[c] * Wl[c * N_CLS + j];
  out[g * N_CLS + j] = acc;
}

extern "C" void kernel_launch(void* const* d_in, const int* in_sizes, int n_in,
                              void* d_out, int out_size, void* d_ws, size_t ws_size,
                              hipStream_t stream) {
  (void)in_sizes; (void)n_in; (void)out_size; (void)ws_size;
  const float* feats = (const float*)d_in[0];
  const int*   src   = (const int*)  d_in[1];
  const int*   dst   = (const int*)  d_in[2];
  const int*   offs  = (const int*)  d_in[3];
  const float* W0  = (const float*)d_in[4];
  const float* b0  = (const float*)d_in[5];
  const float* Wr0 = (const float*)d_in[6];
  const float* br0 = (const float*)d_in[7];
  const float* g0  = (const float*)d_in[8];
  const float* be0 = (const float*)d_in[9];
  const float* W1  = (const float*)d_in[10];
  const float* b1  = (const float*)d_in[11];
  const float* Wr1 = (const float*)d_in[12];
  const float* br1 = (const float*)d_in[13];
  const float* g1  = (const float*)d_in[14];
  const float* be1 = (const float*)d_in[15];
  const float* Wl  = (const float*)d_in[16];
  const float* bl  = (const float*)d_in[17];
  float* out = (float*)d_out;

  // ---- workspace carve-out (256 B aligned) ----
  char* ws = (char*)d_ws;
  size_t off = 0;
  auto alloc = [&](size_t bytes) -> void* {
    void* p = ws + off;
    off = (off + bytes + 255) & ~(size_t)255;
    return p;
  };
  float*    dnOut = (float*)alloc((size_t)N_NODES * 4);
  float*    dnIn  = (float*)alloc((size_t)N_NODES * 4);
  _Float16* W0p   = (_Float16*)alloc((size_t)(IN_F / 32) * 4 * 32 * 16 * 2);
  _Float16* Wr0p  = (_Float16*)alloc((size_t)(IN_F / 32) * 4 * 32 * 16 * 2);
  _Float16* W1p   = (_Float16*)alloc((size_t)(HID / 32) * 4 * 32 * 16 * 2);
  _Float16* Wr1p  = (_Float16*)alloc((size_t)(HID / 32) * 4 * 32 * 16 * 2);
  size_t nodeMatB = (size_t)N_NODES * HID * 4;
  float* B1 = (float*)alloc(nodeMatB);   // gemm out / layer1 gcn raw
  float* B2 = (float*)alloc(nodeMatB);   // agg
  float* B3 = (float*)alloc(nodeMatB);   // residual raw
  float* B4 = (float*)alloc(nodeMatB);   // pre-BN
  float* B5 = (float*)alloc(nodeMatB);   // post-BN h
  float* stats = (float*)alloc(128 * 4);
  float* ss    = (float*)alloc(128 * 4);

  long long nodeElems = (long long)N_NODES * HID;
  int gElem = (int)((nodeElems + 255) / 256);
  long long edgeElems = (long long)N_EDGES * HID;
  int gEdge = (int)((edgeElems + 255) / 256);
  int gemmGrid = (N_NODES + 63) / 64;          // 4 waves/block * 16 rows/wave
  int gNode = (N_NODES + 255) / 256;

  // ---- degrees -> D^-1/2 ----
  zero_f32<<<gNode, 256, 0, stream>>>(dnOut, N_NODES);
  zero_f32<<<gNode, 256, 0, stream>>>(dnIn,  N_NODES);
  deg_kernel<<<(N_EDGES + 255) / 256, 256, 0, stream>>>(src, dst, dnOut, dnIn, N_EDGES);
  dn_finalize<<<gNode, 256, 0, stream>>>(dnOut, dnIn, N_NODES);

  // ---- pack weights into WMMA B-fragments ----
  pack_w<<<4, 128, 0, stream>>>(W0,  W0p,  IN_F);   // 512 fragments*lanes
  pack_w<<<4, 128, 0, stream>>>(Wr0, Wr0p, IN_F);
  pack_w<<<2, 128, 0, stream>>>(W1,  W1p,  HID);    // 256
  pack_w<<<2, 128, 0, stream>>>(Wr1, Wr1p, HID);

  // ================= Layer 0 (IN_F > HID: weight first) =================
  // h0 = (feats * dnOut) @ W0
  gemm_wmma<<<gemmGrid, 128, 0, stream>>>(feats, W0p, dnOut, B1, N_NODES, IN_F);
  // agg = segment_sum(h0[src], dst)
  zero_f32<<<gElem, 256, 0, stream>>>(B2, nodeElems);
  scatter_add<<<gEdge, 256, 0, stream>>>(B1, src, dst, nullptr, B2, N_EDGES);
  // res = feats @ Wr0
  gemm_wmma<<<gemmGrid, 128, 0, stream>>>(feats, Wr0p, nullptr, B3, N_NODES, IN_F);
  // new = gelu(agg*dnIn + b0) + gelu(res + br0)
  epilogue_k<<<gElem, 256, 0, stream>>>(B2, B3, dnIn, b0, br0, B4, N_NODES);
  // BatchNorm
  zero_f32<<<1, 128, 0, stream>>>(stats, 128);
  bn_stats<<<1024, 64, 0, stream>>>(B4, stats, N_NODES);
  bn_finalize<<<1, 64, 0, stream>>>(stats, g0, be0, ss, N_NODES);
  bn_apply<<<gElem, 256, 0, stream>>>(B4, ss, B5, N_NODES);

  // ================= Layer 1 (HID == HID: aggregate first) =================
  // agg1 = segment_sum((h * dnOut)[src], dst)
  zero_f32<<<gElem, 256, 0, stream>>>(B2, nodeElems);
  scatter_add<<<gEdge, 256, 0, stream>>>(B5, src, dst, dnOut, B2, N_EDGES);
  // gcn_raw = (agg1 * dnIn) @ W1
  gemm_wmma<<<gemmGrid, 128, 0, stream>>>(B2, W1p, dnIn, B1, N_NODES, HID);
  // res1 = h @ Wr1
  gemm_wmma<<<gemmGrid, 128, 0, stream>>>(B5, Wr1p, nullptr, B3, N_NODES, HID);
  // new1 = gelu(gcn_raw + b1) + gelu(res1 + br1)   (dnIn already folded)
  epilogue_k<<<gElem, 256, 0, stream>>>(B1, B3, nullptr, b1, br1, B4, N_NODES);
  // BatchNorm
  zero_f32<<<1, 128, 0, stream>>>(stats, 128);
  bn_stats<<<1024, 64, 0, stream>>>(B4, stats, N_NODES);
  bn_finalize<<<1, 64, 0, stream>>>(stats, g1, be1, ss, N_NODES);
  bn_apply<<<gElem, 256, 0, stream>>>(B4, ss, B5, N_NODES);

  // ================= classifier on first node of each graph =================
  classify<<<N_GRAPHS, 64, 0, stream>>>(B5, offs, Wl, bl, out);
}
